// BitSpikeMambaModel_60266981097746
// MI455X (gfx1250) — matrix operand order repaired and checked by reading, not compiled, not run on hardware
//
#include <hip/hip_runtime.h>
#include <hip/hip_bf16.h>

// ---------------------------------------------------------------------------
// BitNet forward on gfx1250 (MI455X): ternary-quantized weights as bf16,
// all three matmuls via v_wmma_f32_16x16x32_bf16.
// Per-wave tile 32(M)x64(N): 8 accumulators, B fragments reused across the
// two M sub-tiles -> 8 WMMAs per 12 b128 loads per K-step.
// ---------------------------------------------------------------------------

typedef __bf16 bf16;
typedef __attribute__((ext_vector_type(16))) __bf16 v16bf;
typedef __attribute__((ext_vector_type(8)))  __bf16 v8bf;
typedef __attribute__((ext_vector_type(8)))  float  v8f;

union AFrag { v16bf v; v8bf h[2]; };

__device__ __forceinline__ v8f wmma_bf16(v16bf a, v16bf b, v8f c) {
  // 8 args: (neg_a, A, neg_b, B, c_mod, C, reuse_a, reuse_b)
  return __builtin_amdgcn_wmma_f32_16x16x32_bf16(false, a, false, b,
                                                 (short)0, c, false, false);
}

#define TOKENS 4096   // B(2) * S(2048)
#define DIM    2048
#define VOCAB  32000
#define RED_BLOCKS 1024

// ---------------------------------------------------------------------------
// Stage 1: per-block partial sums of |w| (deterministic tree reduction)
// ---------------------------------------------------------------------------
__global__ __launch_bounds__(256)
void absmean_partial(const float* __restrict__ w, long long n,
                     float* __restrict__ partial) {
  __shared__ float s[256];
  float acc = 0.f;
  for (long long i = (long long)blockIdx.x * 256 + threadIdx.x; i < n;
       i += (long long)gridDim.x * 256)
    acc += fabsf(w[i]);
  s[threadIdx.x] = acc;
  __syncthreads();
  for (int off = 128; off > 0; off >>= 1) {
    if (threadIdx.x < (unsigned)off) s[threadIdx.x] += s[threadIdx.x + off];
    __syncthreads();
  }
  if (threadIdx.x == 0) partial[blockIdx.x] = s[0];
}

// Stage 2: fixed-order tree over the RED_BLOCKS partials -> scale
__global__ __launch_bounds__(1024)
void finalize_scale(const float* __restrict__ partial, float inv_n,
                    float* __restrict__ scale_out) {
  __shared__ float s[1024];
  s[threadIdx.x] = partial[threadIdx.x];
  __syncthreads();
  for (int off = 512; off > 0; off >>= 1) {
    if (threadIdx.x < (unsigned)off) s[threadIdx.x] += s[threadIdx.x + off];
    __syncthreads();
  }
  if (threadIdx.x == 0) scale_out[0] = fmaxf(s[0] * inv_n, 1e-5f);
}

// ---------------------------------------------------------------------------
// Ternary quantize: wq = bf16(clip(round(w/scale), -1, 1))  (exact in bf16)
// ---------------------------------------------------------------------------
__global__ __launch_bounds__(256)
void quantize_ternary(const float* __restrict__ w,
                      const float* __restrict__ scale_p,
                      bf16* __restrict__ wq, long long n) {
  const float inv = 1.0f / scale_p[0];
  for (long long i = (long long)blockIdx.x * 256 + threadIdx.x; i < n;
       i += (long long)gridDim.x * 256) {
    float q = fminf(fmaxf(rintf(w[i] * inv), -1.0f), 1.0f);
    wq[i] = (bf16)q;
  }
}

// ---------------------------------------------------------------------------
// Embedding gather -> bf16 activations  (one block per token row)
// ---------------------------------------------------------------------------
__global__ __launch_bounds__(256)
void gather_embed(const int* __restrict__ x, const float* __restrict__ emb,
                  bf16* __restrict__ act) {
  const int t = blockIdx.x;
  const long long row = (long long)x[t];
  const float* src = emb + row * DIM;
  bf16* dst = act + (long long)t * DIM;
  for (int d = threadIdx.x; d < DIM; d += 256) dst[d] = (bf16)src[d];
}

// ---------------------------------------------------------------------------
// WMMA GEMM:  out[M,N] = (A[M,K] @ W[N,K]^T) * scale + bias[N]
// 128 threads = 4 waves; wave -> 32x64 tile (8 accumulators), block -> 128x64.
// A fragment (same K order as B): lane m=l&15, 16 contiguous K {0..15} for
// lanes 0-15 and {16..31} for lanes 16-31. B fragment: lane n holds 16
// contiguous K values of W row (col of W^T): kbase 0 (lanes 0-15) / 16 (16-31).
// ---------------------------------------------------------------------------
template <bool OUT_BF16>
__global__ __launch_bounds__(128)
void bitlinear_wmma(const bf16* __restrict__ A, const bf16* __restrict__ W,
                    const float* __restrict__ bias,
                    const float* __restrict__ scale_p,
                    float* __restrict__ outF, bf16* __restrict__ outB,
                    int M, int N, int K) {
  const int lane = threadIdx.x & 31;
  const int wave = threadIdx.x >> 5;
  const int colBase = blockIdx.x * 64;
  const int rowBase = blockIdx.y * 128 + wave * 32;

  const int mloc = lane & 15;
  const int hi = lane >> 4;              // 0 | 1
  const int khalfA = hi * 16;            // A lane K base within 32-slice
  const int khalfB = hi * 16;            // B lane K base within 32-slice

  const bf16* __restrict__ aptr0 = A + (size_t)(rowBase + mloc) * K + khalfA;
  const bf16* __restrict__ aptr1 = aptr0 + (size_t)16 * K;
  const bf16* __restrict__ bptr  = W + (size_t)(colBase + mloc) * K + khalfB;

  v8f acc[8] = {};

  for (int k = 0; k < K; k += 32) {
    __builtin_prefetch(bptr + k + 256, 0, 1);                   // W stream
    __builtin_prefetch(bptr + (size_t)32 * K + k + 256, 0, 1);  // W stream (hi)
    AFrag a0, a1;
    a0.h[0] = *(const v8bf*)(aptr0 + k);
    a0.h[1] = *(const v8bf*)(aptr0 + k + 8);
    a1.h[0] = *(const v8bf*)(aptr1 + k);
    a1.h[1] = *(const v8bf*)(aptr1 + k + 8);
    v16bf b0 = *(const v16bf*)(bptr + k);
    v16bf b1 = *(const v16bf*)(bptr + (size_t)16 * K + k);
    v16bf b2 = *(const v16bf*)(bptr + (size_t)32 * K + k);
    v16bf b3 = *(const v16bf*)(bptr + (size_t)48 * K + k);
    acc[0] = wmma_bf16(a0.v, b0, acc[0]);
    acc[1] = wmma_bf16(a0.v, b1, acc[1]);
    acc[2] = wmma_bf16(a0.v, b2, acc[2]);
    acc[3] = wmma_bf16(a0.v, b3, acc[3]);
    acc[4] = wmma_bf16(a1.v, b0, acc[4]);
    acc[5] = wmma_bf16(a1.v, b1, acc[5]);
    acc[6] = wmma_bf16(a1.v, b2, acc[6]);
    acc[7] = wmma_bf16(a1.v, b3, acc[7]);
  }

  const float scale = scale_p[0];
  // C/D layout: VGPR r -> lanes 0-15: M=r ; lanes 16-31: M=r+8 ; N = lane&15
  const int mOff = hi * 8;
#pragma unroll
  for (int t = 0; t < 2; ++t) {
    const int rowT = rowBase + t * 16;
#pragma unroll
    for (int s = 0; s < 4; ++s) {
      const int n = colBase + s * 16 + mloc;
      const float bn = bias[n];
      const v8f av = acc[t * 4 + s];
#pragma unroll
      for (int r = 0; r < 8; ++r) {
        const int m = rowT + r + mOff;
        const float v = av[r] * scale + bn;
        const size_t idx = (size_t)m * N + n;
        if (OUT_BF16) outB[idx] = (bf16)v;
        else          outF[idx] = v;
      }
    }
  }
}

// ---------------------------------------------------------------------------
// LayerNorm over last dim (2048) -> bf16 activations for the head GEMM
// ---------------------------------------------------------------------------
__global__ __launch_bounds__(256)
void layernorm_to_bf16(const float* __restrict__ y,
                       const float* __restrict__ gamma,
                       const float* __restrict__ beta,
                       bf16* __restrict__ out) {
  __shared__ float s[256];
  __shared__ float mu_s, rstd_s;
  const int row = blockIdx.x;
  const float* yr = y + (long long)row * DIM;

  float acc = 0.f;
  for (int d = threadIdx.x; d < DIM; d += 256) acc += yr[d];
  s[threadIdx.x] = acc;
  __syncthreads();
  for (int off = 128; off > 0; off >>= 1) {
    if (threadIdx.x < (unsigned)off) s[threadIdx.x] += s[threadIdx.x + off];
    __syncthreads();
  }
  if (threadIdx.x == 0) mu_s = s[0] * (1.0f / DIM);
  __syncthreads();
  const float mu = mu_s;

  float v = 0.f;
  for (int d = threadIdx.x; d < DIM; d += 256) {
    float dlt = yr[d] - mu;
    v += dlt * dlt;
  }
  s[threadIdx.x] = v;
  __syncthreads();
  for (int off = 128; off > 0; off >>= 1) {
    if (threadIdx.x < (unsigned)off) s[threadIdx.x] += s[threadIdx.x + off];
    __syncthreads();
  }
  if (threadIdx.x == 0) rstd_s = rsqrtf(s[0] * (1.0f / DIM) + 1e-5f);
  __syncthreads();
  const float rstd = rstd_s;

  bf16* orow = out + (long long)row * DIM;
  for (int d = threadIdx.x; d < DIM; d += 256)
    orow[d] = (bf16)((yr[d] - mu) * rstd * gamma[d] + beta[d]);
}

// ---------------------------------------------------------------------------
// Launch
// ---------------------------------------------------------------------------
extern "C" void kernel_launch(void* const* d_in, const int* in_sizes, int n_in,
                              void* d_out, int out_size, void* d_ws,
                              size_t ws_size, hipStream_t stream) {
  const int*   x     = (const int*)  d_in[0];
  const float* emb   = (const float*)d_in[1];
  const float* w0    = (const float*)d_in[2];
  const float* b0    = (const float*)d_in[3];
  const float* w1    = (const float*)d_in[4];
  const float* b1    = (const float*)d_in[5];
  const float* lng   = (const float*)d_in[6];
  const float* lnb   = (const float*)d_in[7];
  const float* headw = (const float*)d_in[8];
  const float* headb = (const float*)d_in[9];
  float* logits = (float*)d_out;

  // -------- workspace layout (bytes, 256-aligned) --------
  char* ws = (char*)d_ws;
  size_t off = 0;
  auto take = [&](size_t bytes) {
    char* p = ws + off;
    off += (bytes + 255) & ~(size_t)255;
    return p;
  };
  float* partials = (float*)take(3 * RED_BLOCKS * sizeof(float));
  float* scales   = (float*)take(3 * sizeof(float));
  bf16* act0   = (bf16*)take((size_t)TOKENS * DIM * 2);   // 16.8 MB
  bf16* w0q    = (bf16*)take((size_t)DIM * DIM * 2);      //  8.4 MB
  bf16* w1q    = (bf16*)take((size_t)DIM * DIM * 2);      //  8.4 MB
  bf16* headq  = (bf16*)take((size_t)VOCAB * DIM * 2);    // 131 MB (fits L2)
  bf16* act1   = (bf16*)take((size_t)TOKENS * DIM * 2);   // 16.8 MB
  float* y1    = (float*)take((size_t)TOKENS * DIM * 4);  // 33.6 MB
  bf16* actln  = (bf16*)take((size_t)TOKENS * DIM * 2);   // 16.8 MB

  const long long nW  = (long long)DIM * DIM;
  const long long nH  = (long long)VOCAB * DIM;

  // -------- scales (deterministic two-stage reductions) --------
  absmean_partial<<<RED_BLOCKS, 256, 0, stream>>>(w0,    nW, partials + 0 * RED_BLOCKS);
  absmean_partial<<<RED_BLOCKS, 256, 0, stream>>>(w1,    nW, partials + 1 * RED_BLOCKS);
  absmean_partial<<<RED_BLOCKS, 256, 0, stream>>>(headw, nH, partials + 2 * RED_BLOCKS);
  finalize_scale<<<1, 1024, 0, stream>>>(partials + 0 * RED_BLOCKS, 1.0f / (float)nW, scales + 0);
  finalize_scale<<<1, 1024, 0, stream>>>(partials + 1 * RED_BLOCKS, 1.0f / (float)nW, scales + 1);
  finalize_scale<<<1, 1024, 0, stream>>>(partials + 2 * RED_BLOCKS, 1.0f / (float)nH, scales + 2);

  // -------- ternary weights -> bf16 --------
  quantize_ternary<<<2048, 256, 0, stream>>>(w0,    scales + 0, w0q,   nW);
  quantize_ternary<<<2048, 256, 0, stream>>>(w1,    scales + 1, w1q,   nW);
  quantize_ternary<<<8192, 256, 0, stream>>>(headw, scales + 2, headq, nH);

  // -------- embedding gather --------
  gather_embed<<<TOKENS, 256, 0, stream>>>(x, emb, act0);

  // -------- layer 0: [4096,2048] @ [2048,2048]^T -> bf16 --------
  dim3 gblk(128);
  dim3 g0(DIM / 64, TOKENS / 128);
  bitlinear_wmma<true><<<g0, gblk, 0, stream>>>(act0, w0q, b0, scales + 0,
                                                nullptr, act1,
                                                TOKENS, DIM, DIM);
  // -------- layer 1: -> f32 (for layernorm) --------
  bitlinear_wmma<false><<<g0, gblk, 0, stream>>>(act1, w1q, b1, scales + 1,
                                                 y1, nullptr,
                                                 TOKENS, DIM, DIM);
  // -------- layernorm -> bf16 --------
  layernorm_to_bf16<<<TOKENS, 256, 0, stream>>>(y1, lng, lnb, actln);

  // -------- head: [4096,2048] @ [32000,2048]^T -> f32 logits --------
  dim3 gh(VOCAB / 64, TOKENS / 128);
  bitlinear_wmma<false><<<gh, gblk, 0, stream>>>(actln, headq, headb, scales + 2,
                                                 logits, nullptr,
                                                 TOKENS, VOCAB, DIM);
}
